// RegionProposalNetwork_15006615733961
// MI455X (gfx1250) — compile-verified, matrix operand order checked by reading, not run on hardware
//
#include <hip/hip_runtime.h>
#include <hip/hip_bf16.h>
#include <math.h>

// ---------------- problem constants (from reference) ----------------
#define BATCH    8
#define CIN      256
#define HH       50
#define WW       76
#define MID      512
#define HW       (HH*WW)        // 3800
#define KDIM     (CIN*9)        // 2304 (im2col K, order: kpos*256 + c)
#define NTOT     (BATCH*HW)     // 30400 (GEMM N)
#define AANCH    9
#define KANCH    (AANCH*HW)     // 34200 anchors per image
#define PRE_NMS  2000
#define POST_NMS 1000
#define NMS_TH   0.7f

// d_out layout: final (B,1000,4) | objectness (B,9,HW) | bbox_deltas (B,9,4,HW)
#define OUT_FINAL_OFF 0
#define OUT_OBJ_OFF   (BATCH*POST_NMS*4)                 // 32000
#define OUT_REG_OFF   (OUT_OBJ_OFF + BATCH*AANCH*HW)     // 305600

// ---------------- bf16 helpers ----------------
__device__ __forceinline__ unsigned short f32_to_bf16(float f) {
  unsigned int u = __float_as_uint(f);
  u += 0x7FFFu + ((u >> 16) & 1u);   // round-to-nearest-even
  return (unsigned short)(u >> 16);
}
__device__ __forceinline__ float bf16_to_f32(unsigned short s) {
  return __uint_as_float(((unsigned int)s) << 16);
}

// ---------------- WMMA types ----------------
typedef __attribute__((ext_vector_type(16))) __bf16 v16bf;
typedef __attribute__((ext_vector_type(8)))  float  v8f;

union FragBF {
  v16bf v;
  uint4 q[2];
};

// ---------------- optional CDNA5 async global->LDS copy ----------------
#if __has_builtin(__builtin_amdgcn_global_load_async_to_lds_b128)
#define HAVE_ASYNC_LDS 1
typedef int v4i_raw __attribute__((vector_size(16)));
typedef __attribute__((address_space(1))) v4i_raw gv4i;   // global
typedef __attribute__((address_space(3))) v4i_raw lv4i;   // LDS
__device__ __forceinline__ void async_copy16(const void* g, void* l) {
  __builtin_amdgcn_global_load_async_to_lds_b128((gv4i*)g, (lv4i*)l, 0, 0);
}
#else
#define HAVE_ASYNC_LDS 0
#endif

// ---------------- kernel 0a: weights OIHW fp32 -> [m][kpos][c] bf16 ----------------
__global__ void pack_weights_kernel(const float* __restrict__ conv_w,
                                    unsigned short* __restrict__ wpack) {
  int t = blockIdx.x * blockDim.x + threadIdx.x;
  if (t >= MID * KDIM) return;
  int m    = t / KDIM;
  int r    = t - m * KDIM;
  int kpos = r >> 8;       // /256
  int c    = r & 255;
  // conv_w[m][c][kh][kw], flat: (m*CIN + c)*9 + kpos
  wpack[t] = f32_to_bf16(conv_w[(m * CIN + c) * 9 + kpos]);
}

// ---------------- kernel 0b: features NCHW fp32 -> NHWC bf16 ----------------
__global__ void pack_features_kernel(const float* __restrict__ feat,
                                     unsigned short* __restrict__ fpack) {
  int t = blockIdx.x * blockDim.x + threadIdx.x;
  if (t >= BATCH * HW * CIN) return;
  int c    = t & 255;       // CIN == 256
  int rest = t >> 8;
  int hw   = rest % HW;
  int b    = rest / HW;
  fpack[t] = f32_to_bf16(feat[(b * CIN + c) * HW + hw]);
}

// ---------------- kernel 1: conv3x3 + bias + ReLU via WMMA bf16 ----------------
// GEMM: C[m][n] = sum_k A[m][k] * B[k][n],  k = kpos*256 + c
//   A = wpack [512][2304] bf16 ; B = im2col(NHWC features), contiguous in c
// Block tile 128(M) x 64(N); 8 waves: waveM in 0..3 (32 rows), waveN in 0..1 (32 cols).
// K staged 64 at a time -> 8 WMMAs per wave per barrier round.
__global__ __launch_bounds__(256)
void conv3x3_wmma_kernel(const unsigned short* __restrict__ wpack,
                         const unsigned short* __restrict__ fpack,
                         const float* __restrict__ bias,
                         unsigned short* __restrict__ xbf) {
  __shared__ __align__(16) unsigned short sA[128][64];   // [m][k]  16 KB
  __shared__ __align__(16) unsigned short sB[64][64];    // [n][k]   8 KB

  const int tid   = threadIdx.x;
  const int lane  = tid & 31;
  const int wave  = tid >> 5;
  const int waveM = wave & 3;          // 32 rows each
  const int waveN = wave >> 2;         // 32 cols each
  const int tileN = blockIdx.x * 64;   // 475 tiles == 30400 exactly
  const int tileM = blockIdx.y * 128;  // 4 tiles   == 512 exactly

  v8f acc00 = {}, acc01 = {}, acc10 = {}, acc11 = {};

  // A staging: thread t stages 32 bf16 at sA[t/2][(t&1)*32 .. +31]
  const int rowA = tid >> 1;
  const int kkA  = (tid & 1) * 32;
  // B staging: thread t stages 16 bf16 at sB[t/4][(t&3)*16 .. +15]
  const int rowB = tid >> 2;
  const int kkB  = (tid & 3) * 16;

  // decode this thread's staged N coordinate ONCE (no divides in the hot loop)
  const int nG  = tileN + rowB;
  const int bI  = nG / HW;
  const int hwI = nG - bI * HW;
  const int hI  = hwI / WW;
  const int wI  = hwI - hI * WW;

  const unsigned short* aSrc = &wpack[(tileM + rowA) * KDIM + kkA];

  for (int kpos = 0; kpos < 9; ++kpos) {
    const int kh = kpos / 3;
    const int kw = kpos - kh * 3;
    const int ih = hI + kh - 1;
    const int iw = wI + kw - 1;
    const bool valid = ((unsigned)ih < (unsigned)HH) && ((unsigned)iw < (unsigned)WW);
    const unsigned short* bSrc =
        &fpack[(((bI * HH + ih) * WW + iw) * CIN) + kkB];  // only deref'd when valid

    for (int c0 = 0; c0 < CIN; c0 += 64) {
      const unsigned short* as = aSrc + kpos * CIN + c0;
      const unsigned short* bs = bSrc + c0;
#if HAVE_ASYNC_LDS
      // CDNA5 async DMA: memory -> LDS, tracked by ASYNCcnt
#pragma unroll
      for (int q = 0; q < 4; ++q)
        async_copy16(as + q * 8, &sA[rowA][kkA + q * 8]);
      if (valid) {
        async_copy16(bs,     &sB[rowB][kkB]);
        async_copy16(bs + 8, &sB[rowB][kkB + 8]);
      } else {
        uint4 z = make_uint4(0u, 0u, 0u, 0u);
        *(uint4*)&sB[rowB][kkB]     = z;
        *(uint4*)&sB[rowB][kkB + 8] = z;
      }
      asm volatile("s_wait_asynccnt 0" ::: "memory");
#else
      *(uint4*)&sA[rowA][kkA]      = *(const uint4*)(as);
      *(uint4*)&sA[rowA][kkA + 8]  = *(const uint4*)(as + 8);
      *(uint4*)&sA[rowA][kkA + 16] = *(const uint4*)(as + 16);
      *(uint4*)&sA[rowA][kkA + 24] = *(const uint4*)(as + 24);
      uint4 b0v = make_uint4(0u, 0u, 0u, 0u);
      uint4 b1v = b0v;
      if (valid) {
        b0v = *(const uint4*)(bs);
        b1v = *(const uint4*)(bs + 8);
      }
      *(uint4*)&sB[rowB][kkB]     = b0v;
      *(uint4*)&sB[rowB][kkB + 8] = b1v;
#endif
      // prefetch next K-step source lines (gfx1250 global_prefetch_b8)
      if (c0 + 64 < CIN) {
        __builtin_prefetch(as + 64, 0, 0);
        if (valid) __builtin_prefetch(bs + 64, 0, 0);
      }
      __syncthreads();

      // --- fragment loads per ISA 16-bit A/B layout:
      // lanes 0-15: row = lane, K 0..7 then 16..23; lanes 16-31: same row, K+8
      const int kh8 = (lane >> 4) * 8;
      const int mr  = waveM * 32 + (lane & 15);
      const int nr  = waveN * 32 + (lane & 15);
#pragma unroll
      for (int ks = 0; ks < 2; ++ks) {
        const int kb = ks * 32;
        FragBF a0, a1, b0, b1;
        a0.q[0] = *(const uint4*)&sA[mr][kb + kh8];
        a0.q[1] = *(const uint4*)&sA[mr][kb + 16 + kh8];
        a1.q[0] = *(const uint4*)&sA[mr + 16][kb + kh8];
        a1.q[1] = *(const uint4*)&sA[mr + 16][kb + 16 + kh8];
        b0.q[0] = *(const uint4*)&sB[nr][kb + kh8];
        b0.q[1] = *(const uint4*)&sB[nr][kb + 16 + kh8];
        b1.q[0] = *(const uint4*)&sB[nr + 16][kb + kh8];
        b1.q[1] = *(const uint4*)&sB[nr + 16][kb + 16 + kh8];

        acc00 = __builtin_amdgcn_wmma_f32_16x16x32_bf16(false, a0.v, false, b0.v,
                                                        (short)0, acc00, false, false);
        acc01 = __builtin_amdgcn_wmma_f32_16x16x32_bf16(false, a0.v, false, b1.v,
                                                        (short)0, acc01, false, false);
        acc10 = __builtin_amdgcn_wmma_f32_16x16x32_bf16(false, a1.v, false, b0.v,
                                                        (short)0, acc10, false, false);
        acc11 = __builtin_amdgcn_wmma_f32_16x16x32_bf16(false, a1.v, false, b1.v,
                                                        (short)0, acc11, false, false);
      }
      __syncthreads();
    }
  }

  // --- epilogue: C layout -> lanes 0-15: M=r, N=lane ; lanes 16-31: M=8+r.
  // x stored as [b][hw][m] so each fragment row packs to ONE 16B store.
  const int laneHi = (lane >> 4) * 8;
  const int ncol   = lane & 15;
#pragma unroll
  for (int mi = 0; mi < 2; ++mi) {
    const int mBase = tileM + waveM * 32 + mi * 16 + laneHi;
#pragma unroll
    for (int ni = 0; ni < 2; ++ni) {
      const v8f acc = (mi == 0) ? (ni == 0 ? acc00 : acc01)
                                : (ni == 0 ? acc10 : acc11);
      const int n  = tileN + waveN * 32 + ni * 16 + ncol;
      const int b  = n / HW;
      const int hw = n - b * HW;
      __align__(16) unsigned short pk[8];
#pragma unroll
      for (int r = 0; r < 8; ++r) {
        float v = acc[r] + bias[mBase + r];
        v = v > 0.f ? v : 0.f;
        pk[r] = f32_to_bf16(v);
      }
      *(uint4*)&xbf[(size_t)(b * HW + hw) * MID + mBase] = *(const uint4*)pk;
    }
  }
}

// ---------------- kernel 2: 1x1 heads (cls 9ch + reg 36ch) ----------------
__global__ __launch_bounds__(256)
void heads_kernel(const unsigned short* __restrict__ xbf,
                  const float* __restrict__ cls_w, const float* __restrict__ cls_b,
                  const float* __restrict__ reg_w, const float* __restrict__ reg_b,
                  float* __restrict__ obj_out, float* __restrict__ reg_out) {
  int t = blockIdx.x * blockDim.x + threadIdx.x;
  const int total = BATCH * 45 * HW;
  if (t >= total) return;
  int hw   = t % HW;
  int rest = t / HW;
  int o    = rest % 45;
  int b    = rest / 45;
  const float* wrow = (o < 9) ? &cls_w[o * MID] : &reg_w[(o - 9) * MID];
  float acc = (o < 9) ? cls_b[o] : reg_b[o - 9];
  const unsigned short* xr = &xbf[(size_t)(b * HW + hw) * MID];
  for (int c = 0; c < MID; c += 8) {
    uint4 q = *(const uint4*)&xr[c];
    const unsigned short* u = (const unsigned short*)&q;
#pragma unroll
    for (int j = 0; j < 8; ++j)
      acc += bf16_to_f32(u[j]) * wrow[c + j];
  }
  if (o < 9) obj_out[(b * 9 + o) * HW + hw] = acc;
  else       reg_out[(b * 36 + (o - 9)) * HW + hw] = acc;
}

// ---------------- kernel 3: top-2000 per batch (iterative argmax) ----------------
__global__ __launch_bounds__(1024)
void topk_kernel(const float* __restrict__ obj, float* __restrict__ scor,
                 int* __restrict__ tidx) {
  const int b = blockIdx.x, tid = threadIdx.x;
  __shared__ float rv[1024];
  __shared__ int   ri[1024];
  float* s = &scor[b * KANCH];
  const float* src = &obj[b * KANCH];
  for (int k = tid; k < KANCH; k += 1024) s[k] = src[k];
  __syncthreads();
  for (int i = 0; i < PRE_NMS; ++i) {
    float bv = -__builtin_inff();
    int   bi = 0x7fffffff;
    for (int k = tid; k < KANCH; k += 1024) {
      float v = s[k];
      if (v > bv || (v == bv && k < bi)) { bv = v; bi = k; }
    }
    rv[tid] = bv; ri[tid] = bi;
    __syncthreads();
    for (int st = 512; st > 0; st >>= 1) {
      if (tid < st) {
        float v2 = rv[tid + st]; int i2 = ri[tid + st];
        if (v2 > rv[tid] || (v2 == rv[tid] && i2 < ri[tid])) { rv[tid] = v2; ri[tid] = i2; }
      }
      __syncthreads();
    }
    if (tid == 0) { tidx[b * PRE_NMS + i] = ri[0]; s[ri[0]] = -__builtin_inff(); }
    __syncthreads();
  }
}

// ---------------- kernel 4: decode selected anchors ----------------
__global__ __launch_bounds__(256)
void decode_kernel(const int* __restrict__ tidx, const float* __restrict__ reg_out,
                   const int* __restrict__ p_img_h, const int* __restrict__ p_img_w,
                   float* __restrict__ props) {
  int t = blockIdx.x * blockDim.x + threadIdx.x;
  if (t >= BATCH * PRE_NMS) return;
  int b = t / PRE_NMS;
  int k = tidx[t];
  int a  = k / HW;
  int hw = k - a * HW;
  int si = a / 3, rj = a - si * 3;
  float sc = (si == 0) ? 8.f : ((si == 1) ? 16.f : 32.f);
  float rt = (rj == 0) ? 0.5f : ((rj == 1) ? 1.f : 2.f);
  float sq = sqrtf(rt);
  float wa = sc * sq, ha = sc / sq;
  float cx = (float)(hw % WW) * 16.f + 8.f;
  float cy = (float)(hw / WW) * 16.f + 8.f;
  const float* rb = &reg_out[(b * 36 + a * 4) * HW + hw];
  float d0 = rb[0], d1 = rb[HW], d2 = rb[2 * HW], d3 = rb[3 * HW];
  float pcx = d0 * wa + cx;
  float pcy = d1 * ha + cy;
  float pw  = expf(d2) * wa;
  float ph  = expf(d3) * ha;
  float iw = (float)p_img_w[0], ih = (float)p_img_h[0];
  float x1 = (pcx - 0.5f * pw) * iw;
  float y1 = (pcy - 0.5f * ph) * ih;
  float x2 = (pcx + 0.5f * pw) * iw;
  float y2 = (pcy + 0.5f * ph) * ih;
  x1 = fmaxf(x1, 0.f);
  y1 = fmaxf(y1, 0.f);
  x2 = fminf(fmaxf(x2, 0.f), iw);
  y2 = fminf(fmaxf(y2, 0.f), ih);
  float* p = &props[t * 4];
  p[0] = x1; p[1] = y1; p[2] = x2; p[3] = y2;
}

// ---------------- kernel 5: greedy NMS + compact to 1000 ----------------
__global__ __launch_bounds__(256)
void nms_kernel(const float* __restrict__ props, float* __restrict__ final_out) {
  const int b = blockIdx.x, tid = threadIdx.x;
  __shared__ float bx[PRE_NMS][4];       // 32 KB
  __shared__ unsigned char sup[PRE_NMS]; // 2 KB
  __shared__ int ord[POST_NMS];          // 4 KB
  const float* p = &props[b * PRE_NMS * 4];
  for (int i = tid; i < PRE_NMS; i += 256) {
    bx[i][0] = p[i * 4 + 0]; bx[i][1] = p[i * 4 + 1];
    bx[i][2] = p[i * 4 + 2]; bx[i][3] = p[i * 4 + 3];
    sup[i] = 0;
  }
  __syncthreads();
  for (int i = 0; i < PRE_NMS - 1; ++i) {
    if (!sup[i]) {
      const float x1 = bx[i][0], y1 = bx[i][1], x2 = bx[i][2], y2 = bx[i][3];
      const float ai = (x2 - x1) * (y2 - y1);
      for (int j = i + 1 + tid; j < PRE_NMS; j += 256) {
        float xx1 = fmaxf(x1, bx[j][0]);
        float yy1 = fmaxf(y1, bx[j][1]);
        float xx2 = fminf(x2, bx[j][2]);
        float yy2 = fminf(y2, bx[j][3]);
        float inter = fmaxf(xx2 - xx1, 0.f) * fmaxf(yy2 - yy1, 0.f);
        float aj = (bx[j][2] - bx[j][0]) * (bx[j][3] - bx[j][1]);
        float iou = inter / (ai + aj - inter);
        if (iou > NMS_TH) sup[j] = 1;
      }
    }
    __syncthreads();
  }
  if (tid == 0) {
    int cnt = 0;
    for (int i = 0; i < PRE_NMS && cnt < POST_NMS; ++i)
      if (!sup[i]) ord[cnt++] = i;
    for (; cnt < POST_NMS; ++cnt) ord[cnt] = 0;   // pad like where(idx==N, 0, idx)
  }
  __syncthreads();
  for (int j = tid; j < POST_NMS; j += 256) {
    int s = ord[j];
    float* o = &final_out[(b * POST_NMS + j) * 4];
    o[0] = bx[s][0]; o[1] = bx[s][1]; o[2] = bx[s][2]; o[3] = bx[s][3];
  }
}

// ---------------- launcher ----------------
extern "C" void kernel_launch(void* const* d_in, const int* in_sizes, int n_in,
                              void* d_out, int out_size, void* d_ws, size_t ws_size,
                              hipStream_t stream) {
  (void)in_sizes; (void)n_in; (void)out_size; (void)ws_size;
  const float* features = (const float*)d_in[0];
  const float* conv_w   = (const float*)d_in[1];
  const float* conv_b   = (const float*)d_in[2];
  const float* cls_w    = (const float*)d_in[3];
  const float* cls_b    = (const float*)d_in[4];
  const float* reg_w    = (const float*)d_in[5];
  const float* reg_b    = (const float*)d_in[6];
  const int*   img_h    = (const int*)d_in[7];
  const int*   img_w    = (const int*)d_in[8];

  float* out       = (float*)d_out;
  float* final_out = out + OUT_FINAL_OFF;
  float* obj_out   = out + OUT_OBJ_OFF;
  float* reg_out   = out + OUT_REG_OFF;

  // deterministic workspace carve (256B aligned sections)
  char* ws = (char*)d_ws;
  size_t off = 0;
  auto carve = [&](size_t bytes) -> char* {
    char* p = ws + off;
    off = (off + bytes + 255) & ~(size_t)255;
    return p;
  };
  unsigned short* wpack = (unsigned short*)carve((size_t)MID * KDIM * 2);        // 2.36 MB
  unsigned short* fpack = (unsigned short*)carve((size_t)BATCH * HW * CIN * 2);  // 15.6 MB
  unsigned short* xbf   = (unsigned short*)carve((size_t)BATCH * HW * MID * 2);  // 31.1 MB
  float*          scor  = (float*)carve((size_t)BATCH * KANCH * 4);              // 1.1 MB
  int*            tidx  = (int*)carve((size_t)BATCH * PRE_NMS * 4);
  float*          prps  = (float*)carve((size_t)BATCH * PRE_NMS * 4 * 4);

  // 1) layout repack + bf16 conversions
  {
    int nW = MID * KDIM;            // 1,179,648
    pack_weights_kernel<<<(nW + 255) / 256, 256, 0, stream>>>(conv_w, wpack);
    int nF = BATCH * HW * CIN;      // 7,782,400
    pack_features_kernel<<<(nF + 255) / 256, 256, 0, stream>>>(features, fpack);
  }
  // 2) conv3x3 + bias + ReLU (WMMA bf16 GEMM, implicit im2col over NHWC)
  {
    dim3 grid(NTOT / 64, MID / 128);  // (475, 4)
    conv3x3_wmma_kernel<<<grid, 256, 0, stream>>>(wpack, fpack, conv_b, xbf);
  }
  // 3) 1x1 heads -> objectness / bbox_deltas directly in d_out
  {
    int tot = BATCH * 45 * HW;
    heads_kernel<<<(tot + 255) / 256, 256, 0, stream>>>(xbf, cls_w, cls_b,
                                                        reg_w, reg_b, obj_out, reg_out);
  }
  // 4) top-2000 per batch
  topk_kernel<<<BATCH, 1024, 0, stream>>>(obj_out, scor, tidx);
  // 5) decode + scale + clip selected proposals
  {
    int tot = BATCH * PRE_NMS;
    decode_kernel<<<(tot + 255) / 256, 256, 0, stream>>>(tidx, reg_out, img_h, img_w, prps);
  }
  // 6) NMS + compact to final (B,1000,4)
  nms_kernel<<<BATCH, 256, 0, stream>>>(prps, final_out);
}
